// RNN_42356967473616
// MI455X (gfx1250) — matrix operand order repaired and checked by reading, not compiled, not run on hardware
//
#include <hip/hip_runtime.h>
#include <stdint.h>

typedef __attribute__((ext_vector_type(16))) __bf16 v16bf;
typedef __attribute__((ext_vector_type(8)))  float  v8f;
typedef __attribute__((ext_vector_type(4)))  int    v4i;

#define LDSW 40   // 32 bf16 + 8 pad; row = 80B, keeps 16B alignment

#if defined(__gfx1250__) && __has_builtin(__builtin_amdgcn_global_load_async_to_lds_b128)
#define HAVE_ASYNC_LDS 1
#endif

__device__ __forceinline__ uint16_t f2bf(float f) {
  uint32_t u = __float_as_uint(f);
  uint32_t r = u + 0x7FFFu + ((u >> 16) & 1u);   // round-to-nearest-even
  return (uint16_t)(r >> 16);
}

union FragU { uint4 q[2]; v16bf v; };

// A-operand (16x32 bf16): lane L -> row M = mbase + (L&15); half = L>>4.
__device__ __forceinline__ v16bf load_fragA(const uint16_t (*lA)[LDSW], int mbase, int lane) {
  int m = mbase + (lane & 15);
  int h = lane >> 4;
  FragU f;
  f.q[0] = *(const uint4*)&lA[m][8 * h];
  f.q[1] = *(const uint4*)&lA[m][16 + 8 * h];
  return f.v;
}

// B-operand (32x16 bf16) staged transposed as Bt[N][K]; two 16B runs per lane (k=8h.., 16+8h..).
__device__ __forceinline__ v16bf load_fragB(const uint16_t (*lB)[LDSW], int nbase, int lane) {
  int n = nbase + (lane & 15);
  int k0 = (lane >> 4) << 3;
  FragU f;
  f.q[0] = *(const uint4*)&lB[n][k0];
  f.q[1] = *(const uint4*)&lB[n][k0 + 16];
  return f.v;
}

__device__ __forceinline__ v8f wmma_bf16(v16bf a, v16bf b, v8f c) {
  return __builtin_amdgcn_wmma_f32_16x16x32_bf16(false, a, false, b, (short)0, c, false, false);
}

__device__ __forceinline__ void wait_async() {
#ifdef HAVE_ASYNC_LDS
#if __has_builtin(__builtin_amdgcn_s_wait_asynccnt)
  __builtin_amdgcn_s_wait_asynccnt(0);
#else
  asm volatile("s_wait_asynccnt 0x0" ::: "memory");
#endif
#endif
}

// Stage a 128x32 fp32 tile -> bf16 LDS tile (VALU convert path).
__device__ __forceinline__ void stage_f32(uint16_t (*dst)[LDSW], const float* src, int ld) {
  int tid = threadIdx.x;
  int m  = tid >> 1;
  int kk = (tid & 1) << 4;
  const float* p = src + (size_t)m * ld + kk;
  float4 f0 = *(const float4*)(p + 0);
  float4 f1 = *(const float4*)(p + 4);
  float4 f2 = *(const float4*)(p + 8);
  float4 f3 = *(const float4*)(p + 12);
  uint32_t w0 = (uint32_t)f2bf(f0.x) | ((uint32_t)f2bf(f0.y) << 16);
  uint32_t w1 = (uint32_t)f2bf(f0.z) | ((uint32_t)f2bf(f0.w) << 16);
  uint32_t w2 = (uint32_t)f2bf(f1.x) | ((uint32_t)f2bf(f1.y) << 16);
  uint32_t w3 = (uint32_t)f2bf(f1.z) | ((uint32_t)f2bf(f1.w) << 16);
  uint32_t w4 = (uint32_t)f2bf(f2.x) | ((uint32_t)f2bf(f2.y) << 16);
  uint32_t w5 = (uint32_t)f2bf(f2.z) | ((uint32_t)f2bf(f2.w) << 16);
  uint32_t w6 = (uint32_t)f2bf(f3.x) | ((uint32_t)f2bf(f3.y) << 16);
  uint32_t w7 = (uint32_t)f2bf(f3.z) | ((uint32_t)f2bf(f3.w) << 16);
  uint4* q = (uint4*)&dst[m][kk];
  q[0] = make_uint4(w0, w1, w2, w3);
  q[1] = make_uint4(w4, w5, w6, w7);
}

// Stage a 128x32 bf16 tile. Async direct-to-LDS if available, else sync copy.
__device__ __forceinline__ void stage_bf16(uint16_t (*dst)[LDSW], const uint16_t* src, int ld) {
  int tid = threadIdx.x;
  int m  = tid >> 1;
  int kk = (tid & 1) << 4;
  const uint16_t* p = src + (size_t)m * ld + kk;
  uint16_t* l = &dst[m][kk];
#ifdef HAVE_ASYNC_LDS
  typedef __attribute__((address_space(1))) v4i gv4i;   // global int4
  typedef __attribute__((address_space(3))) v4i sv4i;   // LDS int4
  __builtin_amdgcn_global_load_async_to_lds_b128((gv4i*)p, (sv4i*)l, 0, 0);
  __builtin_amdgcn_global_load_async_to_lds_b128((gv4i*)p, (sv4i*)l, 16, 0);
#else
  uint4 a = *(const uint4*)(p);
  uint4 b = *(const uint4*)(p + 8);
  uint4* q = (uint4*)l;
  q[0] = a;
  q[1] = b;
#endif
}

// One K=32 step for a wave's 32x64 sub-tile: 2 A frags x 4 B frags -> 8 WMMA.
__device__ __forceinline__ void mma_step(const uint16_t (*lA)[LDSW], const uint16_t (*lB)[LDSW],
                                         v8f (*acc)[4], int wm, int wn, int lane) {
  v16bf a0 = load_fragA(lA, wm,      lane);
  v16bf a1 = load_fragA(lA, wm + 16, lane);
  v16bf b0 = load_fragB(lB, wn +  0, lane);
  v16bf b1 = load_fragB(lB, wn + 16, lane);
  v16bf b2 = load_fragB(lB, wn + 32, lane);
  v16bf b3 = load_fragB(lB, wn + 48, lane);
  acc[0][0] = wmma_bf16(a0, b0, acc[0][0]);
  acc[0][1] = wmma_bf16(a0, b1, acc[0][1]);
  acc[0][2] = wmma_bf16(a0, b2, acc[0][2]);
  acc[0][3] = wmma_bf16(a0, b3, acc[0][3]);
  acc[1][0] = wmma_bf16(a1, b0, acc[1][0]);
  acc[1][1] = wmma_bf16(a1, b1, acc[1][1]);
  acc[1][2] = wmma_bf16(a1, b2, acc[1][2]);
  acc[1][3] = wmma_bf16(a1, b3, acc[1][3]);
}

// ---------------------------------------------------------------------------
// Kernel 1: H[row, d] = relu( sum_c ins[row, c] * W0[d, c] ), row = t*256+b.
// Software-pipelined, double-buffered LDS, one barrier per K-chunk.
// grid = (8 n-tiles, 512 m-tiles), block = 256 (8 waves, 4x2 wave grid).
// ---------------------------------------------------------------------------
__global__ __launch_bounds__(256) void h_gemm_kernel(
    const float* __restrict__ Mf, const float* __restrict__ DTp,
    const float* __restrict__ Dp, const uint16_t* __restrict__ W0bf,
    uint16_t* __restrict__ Hbf) {
  __shared__ uint16_t lA[2][128][LDSW];
  __shared__ uint16_t lB[2][128][LDSW];
  const int row0 = blockIdx.y * 128;
  const int col0 = blockIdx.x * 128;
  const int lane = threadIdx.x & 31;
  const int wave = threadIdx.x >> 5;
  const int wm = (wave >> 1) * 32;  // 0,32,64,96
  const int wn = (wave & 1) * 64;   // 0,64
  const uint16_t* W0base = W0bf + (size_t)col0 * 1024;

  v8f acc[2][4];
  v8f z = {0.f, 0.f, 0.f, 0.f, 0.f, 0.f, 0.f, 0.f};
#pragma unroll
  for (int i = 0; i < 2; ++i)
#pragma unroll
    for (int j = 0; j < 4; ++j) acc[i][j] = z;

  auto issue_chunk = [&](int kc) {
    int b = kc & 1;
    const float* sA;
    int ld;
    if (kc < 8)       { sA = Mf  + (size_t)row0 * 256 + kc * 32;        ld = 256; }
    else if (kc < 16) { sA = DTp + (size_t)row0 * 256 + (kc - 8) * 32;  ld = 256; }
    else              { sA = Dp  + (size_t)row0 * 512 + (kc - 16) * 32; ld = 512; }
    stage_f32(lA[b], sA, ld);
    stage_bf16(lB[b], W0base + kc * 32, 1024);
  };

  issue_chunk(0);
  for (int kc = 0; kc < 32; ++kc) {
    wait_async();
    __syncthreads();                 // emits s_wait_dscnt 0 + barrier: publishes
    if (kc + 1 < 32) issue_chunk(kc + 1);
    mma_step(lA[kc & 1], lB[kc & 1], acc, wm, wn, lane);
  }

  const int half = lane >> 4;
  const int nl   = lane & 15;
#pragma unroll
  for (int i = 0; i < 2; ++i)
#pragma unroll
    for (int j = 0; j < 4; ++j)
#pragma unroll
      for (int r = 0; r < 8; ++r) {
        float v = fmaxf(acc[i][j][r], 0.0f);
        int row = row0 + wm + i * 16 + 8 * half + r;
        int col = col0 + wn + j * 16 + nl;
        Hbf[(size_t)row * 1024 + col] = f2bf(v);
      }
}

// Agent-scope sense-reversing grid barrier (8 co-resident WGs).
__device__ __forceinline__ void grid_barrier(int* cnt, int* gen, int nwg) {
  __syncthreads();
  __builtin_amdgcn_fence(__ATOMIC_RELEASE, "agent");
  if (threadIdx.x == 0) {
    int g = __hip_atomic_load(gen, __ATOMIC_RELAXED, __HIP_MEMORY_SCOPE_AGENT);
    int v = __hip_atomic_fetch_add(cnt, 1, __ATOMIC_ACQ_REL, __HIP_MEMORY_SCOPE_AGENT);
    if (v == nwg - 1) {
      __hip_atomic_store(cnt, 0, __ATOMIC_RELAXED, __HIP_MEMORY_SCOPE_AGENT);
      __hip_atomic_store(gen, g + 1, __ATOMIC_RELEASE, __HIP_MEMORY_SCOPE_AGENT);
    } else {
      while (__hip_atomic_load(gen, __ATOMIC_ACQUIRE, __HIP_MEMORY_SCOPE_AGENT) == g)
        __builtin_amdgcn_s_sleep(2);
    }
  }
  __syncthreads();
  __builtin_amdgcn_fence(__ATOMIC_ACQUIRE, "agent");
}

// ---------------------------------------------------------------------------
// Kernel 2: persistent scan. grid = 8 WGs tiling the 256x512 state (2x4 tiles
// of 128x128). Per step t: acc = H[t] @ W1^T + x @ Whh^T over 48 pipelined
// K-chunks; relu; write X (f32) + ping-pong bf16 x buffer; grid barrier.
// ---------------------------------------------------------------------------
__global__ __launch_bounds__(256) void rnn_scan_kernel(
    const uint16_t* __restrict__ Hbf, const uint16_t* __restrict__ W1bf,
    const uint16_t* __restrict__ Whhbf, uint16_t* __restrict__ xb0,
    uint16_t* __restrict__ xb1, float* __restrict__ X, int* __restrict__ bar) {
  __shared__ uint16_t lA[2][128][LDSW];
  __shared__ uint16_t lB[2][128][LDSW];
  const int row0 = (blockIdx.x >> 2) * 128;  // b tile
  const int col0 = (blockIdx.x & 3) * 128;   // nx tile
  const int lane = threadIdx.x & 31;
  const int wave = threadIdx.x >> 5;
  const int wm = (wave >> 1) * 32;
  const int wn = (wave & 1) * 64;
  const int half = lane >> 4;
  const int nl   = lane & 15;
  const uint16_t* W1base  = W1bf  + (size_t)col0 * 1024;
  const uint16_t* Whhbase = Whhbf + (size_t)col0 * 512;
  const v8f z = {0.f, 0.f, 0.f, 0.f, 0.f, 0.f, 0.f, 0.f};

  for (int t = 0; t < 256; ++t) {
    uint16_t* xcur = (t & 1) ? xb1 : xb0;
    uint16_t* xnxt = (t & 1) ? xb0 : xb1;
    const uint16_t* Hbase = Hbf + ((size_t)t * 256 + row0) * 1024;

    v8f acc[2][4];
#pragma unroll
    for (int i = 0; i < 2; ++i)
#pragma unroll
      for (int j = 0; j < 4; ++j) acc[i][j] = z;

    auto issue_chunk = [&](int kc) {
      int b = kc & 1;
      if (kc < 32) {                                       // H[t] @ W1^T
        stage_bf16(lA[b], Hbase + kc * 32, 1024);
        stage_bf16(lB[b], W1base + kc * 32, 1024);
      } else {                                             // x @ Whh^T
        int k2 = kc - 32;
        stage_bf16(lA[b], xcur + (size_t)row0 * 512 + k2 * 32, 512);
        stage_bf16(lB[b], Whhbase + k2 * 32, 512);
      }
    };

    issue_chunk(0);
    for (int kc = 0; kc < 48; ++kc) {
      wait_async();
      __syncthreads();
      if (kc + 1 < 48) issue_chunk(kc + 1);
      mma_step(lA[kc & 1], lB[kc & 1], acc, wm, wn, lane);
    }

#pragma unroll
    for (int i = 0; i < 2; ++i)
#pragma unroll
      for (int j = 0; j < 4; ++j)
#pragma unroll
        for (int r = 0; r < 8; ++r) {
          float v = fmaxf(acc[i][j][r], 0.0f);
          int row = row0 + wm + i * 16 + 8 * half + r;
          int col = col0 + wn + j * 16 + nl;
          X[(size_t)t * 131072 + (size_t)row * 512 + col] = v;
          xnxt[(size_t)row * 512 + col] = f2bf(v);
        }

    grid_barrier(bar, bar + 1, (int)gridDim.x);
  }
}

__global__ void cvt_bf16_kernel(const float* __restrict__ src,
                                uint16_t* __restrict__ dst, int n) {
  int i = blockIdx.x * 256 + threadIdx.x;
  if (i < n) dst[i] = f2bf(src[i]);
}

__global__ void init_bar_kernel(int* bar) { bar[0] = 0; bar[1] = 0; }

__global__ void y_gather_kernel(const float* __restrict__ X, float* __restrict__ Y) {
  int i = blockIdx.x * 256 + threadIdx.x;   // 65536 = T*B
  Y[i] = X[(size_t)i * 512 + 511];
}

extern "C" void kernel_launch(void* const* d_in, const int* in_sizes, int n_in,
                              void* d_out, int out_size, void* d_ws, size_t ws_size,
                              hipStream_t stream) {
  const float* x   = (const float*)d_in[0];   // (256, 512)
  const float* Mf  = (const float*)d_in[1];   // (256, 256, 256)
  const float* DTp = (const float*)d_in[2];   // (256, 256, 256)
  const float* Dp  = (const float*)d_in[3];   // (256, 256, 512)
  const float* W0  = (const float*)d_in[4];   // (1024, 1024)
  // d_in[5] = W_hh0: unused by the reference
  const float* W1  = (const float*)d_in[6];   // (512, 1024)
  const float* Whh = (const float*)d_in[7];   // (512, 512)

  float* X = (float*)d_out;                       // (256, 256, 512)
  float* Y = X + (size_t)256 * 256 * 512;         // (256, 256)

  char* ws = (char*)d_ws;
  size_t off = 0;
  uint16_t* Hbf   = (uint16_t*)(ws + off); off += (size_t)65536 * 1024 * 2;  // 128 MB
  uint16_t* W0bf  = (uint16_t*)(ws + off); off += (size_t)1024 * 1024 * 2;
  uint16_t* W1bf  = (uint16_t*)(ws + off); off += (size_t)512 * 1024 * 2;
  uint16_t* Whhbf = (uint16_t*)(ws + off); off += (size_t)512 * 512 * 2;
  uint16_t* xb0   = (uint16_t*)(ws + off); off += (size_t)256 * 512 * 2;
  uint16_t* xb1   = (uint16_t*)(ws + off); off += (size_t)256 * 512 * 2;
  int* bar        = (int*)(ws + off);

  cvt_bf16_kernel<<<(1048576 + 255) / 256, 256, 0, stream>>>(W0, W0bf, 1048576);
  cvt_bf16_kernel<<<(524288 + 255) / 256, 256, 0, stream>>>(W1, W1bf, 524288);
  cvt_bf16_kernel<<<(262144 + 255) / 256, 256, 0, stream>>>(Whh, Whhbf, 262144);
  cvt_bf16_kernel<<<(131072 + 255) / 256, 256, 0, stream>>>(x, xb0, 131072);
  init_bar_kernel<<<1, 1, 0, stream>>>(bar);

  h_gemm_kernel<<<dim3(8, 512), 256, 0, stream>>>(Mf, DTp, Dp, W0bf, Hbf);
  rnn_scan_kernel<<<8, 256, 0, stream>>>(Hbf, W1bf, Whhbf, xb0, xb1, X, bar);
  y_gather_kernel<<<256, 256, 0, stream>>>(X, Y);
}
